// TSGCN_72748156060311
// MI455X (gfx1250) — matrix operand order repaired and checked by reading, not compiled, not run on hardware
//
#include <hip/hip_runtime.h>
#include <hip/hip_bf16.h>

// ---------------------------------------------------------------------------
// Types for CDNA5 WMMA (wave32, 16x16x32 bf16 -> f32)
// ---------------------------------------------------------------------------
typedef __attribute__((ext_vector_type(16))) __bf16 v16bf;
typedef __attribute__((ext_vector_type(8)))  float  v8f;

#define Bdim 32
#define Sdim 512
#define Ndim 64
#define Fdim 16
#define Gdim 64
#define Hdim 256

#define USE_ASYNC_LDS 1  // gfx1250 GLOBAL_LOAD_ASYNC_TO_LDS path in ih_gemm

__device__ __forceinline__ unsigned short f2bf(float f) {
  union { __bf16 h; unsigned short s; } u;
  u.h = (__bf16)f;  // native gfx1250 f32->bf16 convert (RNE)
  return u.s;
}

// 32-bit LDS byte offset of a generic pointer into __shared__ space
__device__ __forceinline__ unsigned lds_addr32(const void* p) {
  return (unsigned)(unsigned long long)(
      __attribute__((address_space(3))) const void*)p;
}

__device__ __forceinline__ v8f wmma_bf16(v16bf a, v16bf b, v8f c) {
  // (neg_a, A, neg_b, B, c_mod, C, reuse_a, reuse_b)
  return __builtin_amdgcn_wmma_f32_16x16x32_bf16(false, a, false, b, (short)0, c,
                                                 false, false);
}

// A fragment: 16x32 bf16 tile, row-major, `tile` points at (m_base, k_base).
// lanes 0-15: M=lane, K=16*hi+0..15, 2 elems per dword.
__device__ __forceinline__ v16bf load_a_frag(const unsigned short* tile, int ldk,
                                             int lane) {
  const int l = lane & 15, hi = lane >> 4;
  const unsigned short* rp = tile + l * ldk;
  union { v16bf v; unsigned u[8]; } f;
#pragma unroll
  for (int j = 0; j < 8; ++j)
    f.u[j] = *(const unsigned*)(rp + 16 * hi + 2 * j);
  return f.v;
}

// B fragment from B^T stored row-major (row = n, ldk along K).
// lanes 0-15: N=lane, K=0..15 (2/dword across 8 VGPRs); lanes 16-31: K=16..31.
__device__ __forceinline__ v16bf load_bt_frag(const unsigned short* tile, int ldk,
                                              int lane) {
  const int l = lane & 15, hi = lane >> 4;
  const unsigned short* rp = tile + l * ldk + 16 * hi;
  union { v16bf v; unsigned u[8]; } f;
#pragma unroll
  for (int j = 0; j < 8; ++j)
    f.u[j] = *(const unsigned*)(rp + 2 * j);
  return f.v;
}

// ---------------------------------------------------------------------------
// Weight prep: f32 -> bf16 copies and small transposes / BN folding
// ---------------------------------------------------------------------------
__global__ void cvt_bf16_kernel(const float* __restrict__ src,
                                unsigned short* __restrict__ dst, int n) {
  int i = blockIdx.x * blockDim.x + threadIdx.x;
  if (i < n) dst[i] = f2bf(src[i]);
}

__global__ void prep_small_kernel(const float* __restrict__ W0,
                                  const float* __restrict__ W1,
                                  const float* __restrict__ bn_g,
                                  const float* __restrict__ bn_b,
                                  const float* __restrict__ bn_m,
                                  const float* __restrict__ bn_v,
                                  unsigned short* __restrict__ W0T,  // 64x32 (g,f pad)
                                  unsigned short* __restrict__ W1T,  // 64x64 (go,gi)
                                  float* __restrict__ bnscale,
                                  float* __restrict__ bnshift) {
  int i = blockIdx.x * blockDim.x + threadIdx.x;
  if (i < 64 * 32) {
    int g = i >> 5, f = i & 31;
    W0T[i] = (f < Fdim) ? f2bf(W0[f * Gdim + g]) : (unsigned short)0;
  }
  int j = i - 64 * 32;
  if (j >= 0 && j < 64 * 64) {
    int go = j >> 6, gi = j & 63;
    W1T[j] = f2bf(W1[gi * Gdim + go]);
  }
  int k = i - (64 * 32 + 64 * 64);
  if (k >= 0 && k < 64) {
    float inv = rsqrtf(bn_v[k] + 1e-5f);
    float sc = inv * bn_g[k];
    bnscale[k] = sc;
    bnshift[k] = bn_b[k] - bn_m[k] * sc;
  }
}

// ---------------------------------------------------------------------------
// GCN: per (b,s): H2 = relu(adj @ BN(relu(adj@x@W0+b0)) @ W1 + b1) -> seq bf16
// One workgroup (4 waves) per (b,s); all GEMMs via WMMA on LDS tiles.
// ---------------------------------------------------------------------------
__global__ __launch_bounds__(128) void gcn_kernel(
    const float* __restrict__ x, const unsigned short* __restrict__ adj_bf,
    const unsigned short* __restrict__ W0T_bf,
    const unsigned short* __restrict__ W1T_bf, const float* __restrict__ bnscale,
    const float* __restrict__ bnshift, const float* __restrict__ b0,
    const float* __restrict__ b1, unsigned short* __restrict__ seq_out) {
  __shared__ unsigned short s_adj[64 * 64];  // A (u,v)
  __shared__ unsigned short s_xT[16 * 64];   // B^T for L1a (f,v)
  __shared__ unsigned short s_T[64 * 32];    // A for L1b (u, f pad 32)
  __shared__ unsigned short s_W0T[64 * 32];  // B^T for L1b (g, f pad)
  __shared__ unsigned short s_H1T[64 * 64];  // B^T for L2a (g, v)
  __shared__ unsigned short s_T2[64 * 64];   // A for L2b (u, g)
  __shared__ unsigned short s_W1T[64 * 64];  // B^T for L2b (go, gi)
  __shared__ float s_bns[64], s_bnb[64], s_b0[64], s_b1[64];

  const int tid = threadIdx.x;
  const int lane = tid & 31;
  const int w = tid >> 5;  // 0..3
  const int bs = blockIdx.x;
  const int l = lane & 15, hi = lane >> 4;

  for (int i = tid; i < 64 * 64; i += 128) {
    s_adj[i] = adj_bf[i];
    s_W1T[i] = W1T_bf[i];
  }
  for (int i = tid; i < 64 * 32; i += 128) s_W0T[i] = W0T_bf[i];
  if (tid < 64) {
    s_bns[tid] = bnscale[tid];
    s_bnb[tid] = bnshift[tid];
    s_b0[tid] = b0[tid];
    s_b1[tid] = b1[tid];
  }
  const float* xp = x + (size_t)bs * (Ndim * Fdim);
  for (int i = tid; i < Ndim * Fdim; i += 128) {
    int v = i >> 4, f = i & 15;
    s_xT[f * 64 + v] = f2bf(xp[i]);  // transpose into B^T layout
  }
  for (int i = tid; i < 64 * 16; i += 128) {  // zero K pad of s_T (cols 16..31)
    int r = i >> 4, c = i & 15;
    s_T[r * 32 + 16 + c] = 0;
  }
  __syncthreads();

  const int m_base = 16 * w;

  // ---- L1a: T = adj @ x  (64x16, K=64) ----
  {
    v8f acc = {};
#pragma unroll
    for (int kk = 0; kk < 64; kk += 32) {
      v16bf a = load_a_frag(s_adj + m_base * 64 + kk, 64, lane);
      v16bf bt = load_bt_frag(s_xT + kk, 64, lane);
      acc = wmma_bf16(a, bt, acc);
    }
#pragma unroll
    for (int r = 0; r < 8; ++r) {
      int m = m_base + r + 8 * hi;
      s_T[m * 32 + l] = f2bf(acc[r]);
    }
  }
  __syncthreads();

  // ---- L1b: H1 = BN(relu(T@W0+b0)), stored transposed (g, u) ----
  {
    v16bf a = load_a_frag(s_T + m_base * 32, 32, lane);
#pragma unroll
    for (int nt = 0; nt < 4; ++nt) {
      int n_base = nt * 16;
      float bb = s_b0[n_base + l];
      v8f acc = {bb, bb, bb, bb, bb, bb, bb, bb};
      v16bf bt = load_bt_frag(s_W0T + n_base * 32, 32, lane);
      acc = wmma_bf16(a, bt, acc);
#pragma unroll
      for (int r = 0; r < 8; ++r) {
        int m = m_base + r + 8 * hi;
        float v = acc[r];
        v = v > 0.f ? v : 0.f;
        v = v * s_bns[m] + s_bnb[m];  // BN over node dim (rows)
        s_H1T[(n_base + l) * 64 + m] = f2bf(v);
      }
    }
  }
  __syncthreads();

  // ---- L2a: T2 = adj @ H1 (64x64, K=64) ----
  {
#pragma unroll
    for (int nt = 0; nt < 4; ++nt) {
      int n_base = nt * 16;
      v8f acc = {};
#pragma unroll
      for (int kk = 0; kk < 64; kk += 32) {
        v16bf a = load_a_frag(s_adj + m_base * 64 + kk, 64, lane);
        v16bf bt = load_bt_frag(s_H1T + n_base * 64 + kk, 64, lane);
        acc = wmma_bf16(a, bt, acc);
      }
#pragma unroll
      for (int r = 0; r < 8; ++r) {
        int m = m_base + r + 8 * hi;
        s_T2[m * 64 + n_base + l] = f2bf(acc[r]);
      }
    }
  }
  __syncthreads();

  // ---- L2b: H2 = relu(T2@W1+b1) -> global seq (b*S+s, u*64+g) bf16 ----
  {
    unsigned short* outp = seq_out + (size_t)bs * (Ndim * Gdim);
#pragma unroll
    for (int nt = 0; nt < 4; ++nt) {
      int n_base = nt * 16;
      float bb = s_b1[n_base + l];
      v8f acc = {bb, bb, bb, bb, bb, bb, bb, bb};
#pragma unroll
      for (int kk = 0; kk < 64; kk += 32) {
        v16bf a = load_a_frag(s_T2 + m_base * 64 + kk, 64, lane);
        v16bf bt = load_bt_frag(s_W1T + n_base * 64 + kk, 64, lane);
        acc = wmma_bf16(a, bt, acc);
      }
#pragma unroll
      for (int r = 0; r < 8; ++r) {
        int m = m_base + r + 8 * hi;
        float v = acc[r];
        outp[m * 64 + n_base + l] = f2bf(v > 0.f ? v : 0.f);
      }
    }
  }
}

// ---------------------------------------------------------------------------
// Batched input-to-hidden GEMM: C[16384,768] = A[16384,K] @ W[768,K]^T + bias
// 256 threads (8 waves). Workgroup tile 128x128; wave = 1 M-tile x 8 N-tiles.
// Double-buffered LDS staging via GLOBAL_LOAD_ASYNC_TO_LDS_B128 (ASYNCcnt):
// the copy for K-chunk k+1 overlaps the WMMAs on chunk k; s_wait_asynccnt 4
// waits only for the oldest 4 async ops (async loads complete in order).
// ---------------------------------------------------------------------------
__global__ __launch_bounds__(256) void ih_gemm_kernel(
    const unsigned short* __restrict__ A, const unsigned short* __restrict__ W,
    const float* __restrict__ bias, float* __restrict__ C, int K, int Ntot) {
  __shared__ unsigned short sA[2][128 * 32];
  __shared__ unsigned short sB[2][128 * 32];
  const int tid = threadIdx.x, lane = tid & 31, w = tid >> 5;
  const int m0 = blockIdx.x * 128;
  const int n0 = blockIdx.y * 128;

  v8f acc[8];
#pragma unroll
  for (int j = 0; j < 8; ++j) acc[j] = (v8f){0, 0, 0, 0, 0, 0, 0, 0};

  const int lrow = tid >> 1;        // 0..127
  const int lcol = (tid & 1) * 16;  // 0 or 16
  const unsigned short* Arow = A + (size_t)(m0 + lrow) * K + lcol;
  const unsigned short* Wrow = W + (size_t)(n0 + lrow) * K + lcol;
  const unsigned soff = lrow * 32 + lcol;

#if USE_ASYNC_LDS
  const unsigned laA[2] = {lds_addr32(&sA[0][soff]), lds_addr32(&sA[1][soff])};
  const unsigned laB[2] = {lds_addr32(&sB[0][soff]), lds_addr32(&sB[1][soff])};

  // Issue 4 async b128 copies (32B of A-row + 32B of W-row) into buffer `buf`.
  // The instruction offset applies to BOTH the LDS and global address.
  auto stage = [&](int buf, int kk) {
    unsigned la = laA[buf], lb = laB[buf];
    unsigned long long ga = (unsigned long long)(Arow + kk);
    unsigned long long gb = (unsigned long long)(Wrow + kk);
    asm volatile(
        "global_load_async_to_lds_b128 %0, %2, off\n\t"
        "global_load_async_to_lds_b128 %0, %2, off offset:16\n\t"
        "global_load_async_to_lds_b128 %1, %3, off\n\t"
        "global_load_async_to_lds_b128 %1, %3, off offset:16"
        :
        : "v"(la), "v"(lb), "v"(ga), "v"(gb)
        : "memory");
  };

  stage(0, 0);
  for (int kk = 0; kk < K; kk += 32) {
    const int buf = (kk >> 5) & 1;
    if (kk + 32 < K) {
      stage(buf ^ 1, kk + 32);  // prefetch next chunk into other buffer
      asm volatile("s_wait_asynccnt 0x4" ::: "memory");  // wait current chunk
    } else {
      asm volatile("s_wait_asynccnt 0x0" ::: "memory");
    }
    __syncthreads();  // all lanes' async data for `buf` visible
    v16bf af = load_a_frag(&sA[buf][(16 * w) * 32], 32, lane);
#pragma unroll
    for (int j = 0; j < 8; ++j) {
      v16bf bf = load_bt_frag(&sB[buf][(16 * j) * 32], 32, lane);
      acc[j] = wmma_bf16(af, bf, acc[j]);
    }
    __syncthreads();  // reads of `buf` done before it is restaged next iter
  }
#else
  unsigned short* sArow = &sA[0][soff];
  unsigned short* sBrow = &sB[0][soff];
  for (int kk = 0; kk < K; kk += 32) {
    const uint4* ap = (const uint4*)(Arow + kk);
    const uint4* wp = (const uint4*)(Wrow + kk);
    uint4 a0 = ap[0], a1 = ap[1];
    uint4 b0v = wp[0], b1v = wp[1];
    if (kk + 32 < K) {
      __builtin_prefetch(Arow + kk + 32, 0, 1);
      __builtin_prefetch(Wrow + kk + 32, 0, 1);
    }
    __syncthreads();
    *(uint4*)(sArow) = a0;
    *(uint4*)(sArow + 8) = a1;
    *(uint4*)(sBrow) = b0v;
    *(uint4*)(sBrow + 8) = b1v;
    __syncthreads();
    v16bf af = load_a_frag(&sA[0][(16 * w) * 32], 32, lane);
#pragma unroll
    for (int j = 0; j < 8; ++j) {
      v16bf bf = load_bt_frag(&sB[0][(16 * j) * 32], 32, lane);
      acc[j] = wmma_bf16(af, bf, acc[j]);
    }
  }
#endif

  const int l = lane & 15, hi = lane >> 4;
#pragma unroll
  for (int j = 0; j < 8; ++j) {
    int n = n0 + 16 * j + l;
    float bv = bias[n];
#pragma unroll
    for (int r = 0; r < 8; ++r) {
      int m = m0 + 16 * w + r + 8 * hi;
      C[(size_t)m * Ntot + n] = acc[j][r] + bv;
    }
  }
}

// ---------------------------------------------------------------------------
// GRU recurrence, single workgroup of 32 waves (1024 threads).
// Each wave owns (batch-tile mt, hidden-tile jt) and computes all 3 gate
// slices (n, 256+n, 512+n) so gates update entirely in registers.
// gi rows are indexed (b*S + t) to match the ih GEMM layout.
// ---------------------------------------------------------------------------
__global__ __launch_bounds__(1024) void gru_kernel(
    const float* __restrict__ gi, const unsigned short* __restrict__ Whh,
    const float* __restrict__ bhh, unsigned short* __restrict__ y_out,
    float* __restrict__ h_final, int S_steps) {
  __shared__ unsigned short s_h[Bdim * Hdim];  // bf16 h for WMMA A operand
  __shared__ float s_hf[Bdim * Hdim];          // f32 master copy of h

  const int tid = threadIdx.x, lane = tid & 31, w = tid >> 5;  // w 0..31
  const int mt = w & 1;   // batch tile (0-15 / 16-31)
  const int jt = w >> 1;  // hidden tile 0..15
  const int l = lane & 15, hi = lane >> 4;

  for (int i = tid; i < Bdim * Hdim; i += 1024) {
    s_h[i] = 0;
    s_hf[i] = 0.f;
  }
  __syncthreads();

  for (int t = 0; t < S_steps; ++t) {
    v8f accs[3];
#pragma unroll
    for (int g = 0; g < 3; ++g) {
      int n_base = g * Hdim + jt * 16;
      float bb = bhh[n_base + l];
      v8f acc = {bb, bb, bb, bb, bb, bb, bb, bb};
#pragma unroll
      for (int kk = 0; kk < Hdim; kk += 32) {
        v16bf af = load_a_frag(s_h + (16 * mt) * Hdim + kk, Hdim, lane);
        v16bf bf = load_bt_frag(Whh + (size_t)n_base * Hdim + kk, Hdim, lane);
        acc = wmma_bf16(af, bf, acc);
      }
      accs[g] = acc;
    }
    __syncthreads();  // all waves done reading s_h

#pragma unroll
    for (int r = 0; r < 8; ++r) {
      int b = 16 * mt + r + 8 * hi;
      int j = jt * 16 + l;
      const float* gir = gi + ((size_t)b * S_steps + t) * 768;
      float g_r = gir[j] + accs[0][r];
      float g_z = gir[256 + j] + accs[1][r];
      float g_n = gir[512 + j];
      float rg = 1.f / (1.f + __expf(-g_r));
      float zg = 1.f / (1.f + __expf(-g_z));
      float ng = tanhf(g_n + rg * accs[2][r]);
      float hold = s_hf[b * Hdim + j];
      float hnew = (1.f - zg) * ng + zg * hold;
      s_hf[b * Hdim + j] = hnew;
      s_h[b * Hdim + j] = f2bf(hnew);
      if (y_out) y_out[((size_t)b * S_steps + t) * Hdim + j] = f2bf(hnew);
    }
    __syncthreads();  // h updates visible before next step's GEMM
  }

  if (h_final) {
    for (int i = tid; i < Bdim * Hdim; i += 1024) h_final[i] = s_hf[i];
  }
}

// ---------------------------------------------------------------------------
// Head: out = relu(h_last @ fc1 + b1f) @ fc2 + b2f   (tiny, scalar f32)
// ---------------------------------------------------------------------------
__global__ __launch_bounds__(256) void head_kernel(
    const float* __restrict__ hlast, const float* __restrict__ fc1_w,
    const float* __restrict__ fc1_b, const float* __restrict__ fc2_w,
    const float* __restrict__ fc2_b, float* __restrict__ out) {
  __shared__ float s_h[32 * 256];
  __shared__ float s_o1[32 * 128];
  const int tid = threadIdx.x;
  for (int i = tid; i < 32 * 256; i += 256) s_h[i] = hlast[i];
  __syncthreads();
  for (int i = tid; i < 32 * 128; i += 256) {
    int b = i >> 7, n = i & 127;
    float acc = fc1_b[n];
    for (int k = 0; k < 256; ++k) acc += s_h[b * 256 + k] * fc1_w[k * 128 + n];
    s_o1[i] = acc > 0.f ? acc : 0.f;
  }
  __syncthreads();
  for (int i = tid; i < 32 * 64; i += 256) {
    int b = i >> 6, n = i & 63;
    float acc = fc2_b[n];
    for (int k = 0; k < 128; ++k) acc += s_o1[b * 128 + k] * fc2_w[k * 64 + n];
    out[i] = acc;
  }
}

// ---------------------------------------------------------------------------
// Host orchestration
// ---------------------------------------------------------------------------
extern "C" void kernel_launch(void* const* d_in, const int* in_sizes, int n_in,
                              void* d_out, int out_size, void* d_ws,
                              size_t ws_size, hipStream_t stream) {
  const float* x    = (const float*)d_in[0];
  const float* adj  = (const float*)d_in[1];
  const float* W0   = (const float*)d_in[2];
  const float* b0   = (const float*)d_in[3];
  const float* W1   = (const float*)d_in[4];
  const float* b1   = (const float*)d_in[5];
  const float* bn_g = (const float*)d_in[6];
  const float* bn_b = (const float*)d_in[7];
  const float* bn_m = (const float*)d_in[8];
  const float* bn_v = (const float*)d_in[9];
  const float* Wih0 = (const float*)d_in[10];
  const float* Whh0 = (const float*)d_in[11];
  const float* bih0 = (const float*)d_in[12];
  const float* bhh0 = (const float*)d_in[13];
  const float* Wih1 = (const float*)d_in[14];
  const float* Whh1 = (const float*)d_in[15];
  const float* bih1 = (const float*)d_in[16];
  const float* bhh1 = (const float*)d_in[17];
  const float* fc1w = (const float*)d_in[18];
  const float* fc1b = (const float*)d_in[19];
  const float* fc2w = (const float*)d_in[20];
  const float* fc2b = (const float*)d_in[21];
  float* out = (float*)d_out;

  char* ws = (char*)d_ws;
  size_t off = 0;
  auto alloc = [&](size_t bytes) -> void* {
    void* p = ws + off;
    off += (bytes + 255) & ~(size_t)255;
    return p;
  };

  unsigned short* adj_bf  = (unsigned short*)alloc((size_t)64 * 64 * 2);
  unsigned short* W0T_bf  = (unsigned short*)alloc((size_t)64 * 32 * 2);
  unsigned short* W1T_bf  = (unsigned short*)alloc((size_t)64 * 64 * 2);
  float*          bnscale = (float*)alloc((size_t)64 * 4);
  float*          bnshift = (float*)alloc((size_t)64 * 4);
  unsigned short* Wih0_bf = (unsigned short*)alloc((size_t)768 * 4096 * 2);
  unsigned short* Whh0_bf = (unsigned short*)alloc((size_t)768 * 256 * 2);
  unsigned short* Wih1_bf = (unsigned short*)alloc((size_t)768 * 256 * 2);
  unsigned short* Whh1_bf = (unsigned short*)alloc((size_t)768 * 256 * 2);
  unsigned short* seq0    = (unsigned short*)alloc((size_t)16384 * 4096 * 2);
  float*          gi_buf  = (float*)alloc((size_t)16384 * 768 * 4);
  unsigned short* y0      = (unsigned short*)alloc((size_t)16384 * 256 * 2);
  float*          hlast   = (float*)alloc((size_t)32 * 256 * 4);

  // --- weight conversion ---
  cvt_bf16_kernel<<<(64 * 64 + 255) / 256, 256, 0, stream>>>(adj, adj_bf, 64 * 64);
  cvt_bf16_kernel<<<(768 * 4096 + 255) / 256, 256, 0, stream>>>(Wih0, Wih0_bf,
                                                                768 * 4096);
  cvt_bf16_kernel<<<(768 * 256 + 255) / 256, 256, 0, stream>>>(Whh0, Whh0_bf,
                                                               768 * 256);
  cvt_bf16_kernel<<<(768 * 256 + 255) / 256, 256, 0, stream>>>(Wih1, Wih1_bf,
                                                               768 * 256);
  cvt_bf16_kernel<<<(768 * 256 + 255) / 256, 256, 0, stream>>>(Whh1, Whh1_bf,
                                                               768 * 256);
  prep_small_kernel<<<(64 * 32 + 64 * 64 + 64 + 255) / 256, 256, 0, stream>>>(
      W0, W1, bn_g, bn_b, bn_m, bn_v, W0T_bf, W1T_bf, bnscale, bnshift);

  // --- GCN stage: one workgroup per (b,s) ---
  gcn_kernel<<<Bdim * Sdim, 128, 0, stream>>>(x, adj_bf, W0T_bf, W1T_bf, bnscale,
                                              bnshift, b0, b1, seq0);

  // --- GRU layer 0: batched ih GEMM then recurrence ---
  ih_gemm_kernel<<<dim3(16384 / 128, 768 / 128), 256, 0, stream>>>(
      seq0, Wih0_bf, bih0, gi_buf, 4096, 768);
  gru_kernel<<<1, 1024, 0, stream>>>(gi_buf, Whh0_bf, bhh0, y0, nullptr, Sdim);

  // --- GRU layer 1 (gi buffer reused) ---
  ih_gemm_kernel<<<dim3(16384 / 128, 768 / 128), 256, 0, stream>>>(
      y0, Wih1_bf, bih1, gi_buf, 256, 768);
  gru_kernel<<<1, 1024, 0, stream>>>(gi_buf, Whh1_bf, bhh1, nullptr, hlast, Sdim);

  // --- FC head ---
  head_kernel<<<1, 256, 0, stream>>>(hlast, fc1w, fc1b, fc2w, fc2b, out);
}
